// GAT_36000415875158
// MI455X (gfx1250) — matrix operand order repaired and checked by reading, not compiled, not run on hardware
//
#include <hip/hip_runtime.h>
#include <hip/hip_bf16.h>
#include <cstdint>

// ---------------------------------------------------------------------------
// GAT (3 layers) + mean pool + linear, for gfx1250 (CDNA5, wave32, WMMA).
// GEMMs: f16 WMMA (v_wmma_f32_16x16x32_f16), f32 accumulate, double-buffered
// LDS fed by CDNA5 async global->LDS DMA (global_load_async_to_lds_b128,
// tracked with ASYNCcnt). Edge softmax/aggregation: fp32 atomics (L2-resident).
// ---------------------------------------------------------------------------

typedef __attribute__((ext_vector_type(16))) _Float16 v16h;
typedef __attribute__((ext_vector_type(8)))  float    v8f;

struct F16x16Bits { uint4 lo, hi; };   // 32 bytes == v16h

#define BM 128
#define BN 128
#define BK 32

// --- CDNA5 async global->LDS (inline asm; tracked by ASYNCcnt) -------------
__device__ __forceinline__ void async_load_b128(const void* gptr, const void* lds_generic)
{
    unsigned           lds = (unsigned)(uintptr_t)lds_generic;       // AS3 offset
    unsigned long long ga  = (unsigned long long)(uintptr_t)gptr;    // 64-bit VA
    asm volatile("global_load_async_to_lds_b128 %0, %1, off"
                 :: "v"(lds), "v"(ga) : "memory");
}
__device__ __forceinline__ void wait_async0()
{
    asm volatile("s_wait_asynccnt 0x0" ::: "memory");
}
__device__ __forceinline__ void wait_ds0()
{
    asm volatile("s_wait_dscnt 0x0" ::: "memory");
}

// ---------------------------------------------------------------------------
// WMMA GEMM: C[N,M] = A[N,K] (f16) * Bt[M,K]^T (f16, pre-transposed weights)
// Block: 256 threads = 8 waves; block tile 128x128; wave (wr,wc) in 2x4 grid
// owns a 64x32 quadrant = 4x2 WMMA tiles = 8 v_wmma per k-step.
// LDS double-buffered, filled by async DMA of the next k-tile.
// ---------------------------------------------------------------------------
__global__ __launch_bounds__(256)
void gat_wmma_gemm_f16(const _Float16* __restrict__ A,
                       const _Float16* __restrict__ Bt,
                       float* __restrict__ C,
                       int N, int K, int M)
{
    __shared__ __align__(16) _Float16 AsBuf[2][BM * BK];  // 2 x 8 KB
    __shared__ __align__(16) _Float16 BsBuf[2][BN * BK];  // 2 x 8 KB

    const int tid   = threadIdx.x;
    const int wave  = tid >> 5;
    const int lane  = tid & 31;
    const int wr    = wave >> 2;        // 0..1  -> row quadrant (64 rows)
    const int wc    = wave & 3;         // 0..3  -> col quadrant (32 cols)
    const int row0  = blockIdx.x * BM;
    const int col0  = blockIdx.y * BN;

    const int r     = lane & 15;        // index within 16x16 tile
    const int khalf = lane >> 4;        // 0: lanes 0-15, 1: lanes 16-31

    // per-thread staging chunks (16B each): 512 chunks per tile, 2 per thread
    const int c0row = tid >> 2;                 // chunk row (0..63 for tid<256? no: 0..63?) -> tid/4: 0..63
    const int c0col = (tid & 3) << 3;           // chunk col element (0,8,16,24)
    // chunk ids tid and tid+256 -> rows tid/4 and tid/4+64

    v8f acc[4][2];
    #pragma unroll
    for (int i = 0; i < 4; ++i)
        #pragma unroll
        for (int j = 0; j < 2; ++j)
            acc[i][j] = v8f{};

    auto issue_tile = [&](int buf, int k0) {
        // A tile: rows row0..row0+127, cols k0..k0+31
        #pragma unroll
        for (int i = 0; i < 2; ++i) {
            int arow = c0row + 64 * i;
            int grow = row0 + arow;
            if (grow < N)
                async_load_b128(A + (size_t)grow * K + k0 + c0col,
                                &AsBuf[buf][arow * BK + c0col]);
        }
        // B tile: Bt rows col0..col0+127, cols k0..k0+31
        #pragma unroll
        for (int i = 0; i < 2; ++i) {
            int brow = c0row + 64 * i;
            int gn = col0 + brow;
            if (gn < M)
                async_load_b128(Bt + (size_t)gn * K + k0 + c0col,
                                &BsBuf[buf][brow * BK + c0col]);
        }
    };

    const int nk = K / BK;
    issue_tile(0, 0);

    for (int i = 0; i < nk; ++i) {
        const int cur = i & 1;
        // my async loads for buf[cur] landed + my LDS reads of last iter done
        wait_async0();
        wait_ds0();
        __syncthreads();

        if (i + 1 < nk) issue_tile(cur ^ 1, (i + 1) * BK);

        const _Float16* As = AsBuf[cur];
        const _Float16* Bs = BsBuf[cur];

        // B fragments (2): n = 32*wc + 16*tj + r; lanes 0-15 K{0-7,16-23}, 16-31 K{8-15,24-31}
        v16h bfrag[2];
        #pragma unroll
        for (int tj = 0; tj < 2; ++tj) {
            const _Float16* bp = &Bs[(32 * wc + 16 * tj + r) * BK + khalf * 8];
            F16x16Bits bits;
            bits.lo = *(const uint4*)(bp);
            bits.hi = *(const uint4*)(bp + 16);
            bfrag[tj] = __builtin_bit_cast(v16h, bits);
        }
        // A fragments (4): row = 64*wr + 16*ti + r; lanes 0-15 K{0-7,16-23},
        // lanes 16-31 K{8-15,24-31}
        #pragma unroll
        for (int ti = 0; ti < 4; ++ti) {
            const _Float16* ap = &As[(64 * wr + 16 * ti + r) * BK + khalf * 8];
            F16x16Bits bits;
            bits.lo = *(const uint4*)(ap);
            bits.hi = *(const uint4*)(ap + 16);
            v16h afrag = __builtin_bit_cast(v16h, bits);
            #pragma unroll
            for (int tj = 0; tj < 2; ++tj)
                acc[ti][tj] = __builtin_amdgcn_wmma_f32_16x16x32_f16(
                    false, afrag, false, bfrag[tj], (short)0, acc[ti][tj],
                    false, false);
        }
    }

    // C/D layout: VGPR j -> (m = khalf*8 + j, n = r) within each 16x16 tile.
    const int crow_base = row0 + 64 * wr + khalf * 8;
    #pragma unroll
    for (int ti = 0; ti < 4; ++ti) {
        #pragma unroll
        for (int tj = 0; tj < 2; ++tj) {
            int cc = col0 + 32 * wc + 16 * tj + r;
            if (cc >= M) continue;
            #pragma unroll
            for (int j = 0; j < 8; ++j) {
                int rr = crow_base + 16 * ti + j;
                if (rr < N) C[(size_t)rr * M + cc] = acc[ti][tj][j];
            }
        }
    }
}

// ---------------------------------------------------------------------------
// Helpers / elementwise kernels
// ---------------------------------------------------------------------------
__global__ void k_fill_f32(float* __restrict__ p, float v, long long n)
{
    long long i = blockIdx.x * 256LL + threadIdx.x;
    if (i < n) p[i] = v;
}

__global__ void k_f32_to_f16(const float* __restrict__ in, _Float16* __restrict__ out,
                             long long n)
{
    long long i = blockIdx.x * 256LL + threadIdx.x;
    if (i < n) out[i] = (_Float16)in[i];
}

// Wt[m][k] = (f16) W[k][m]   (W is [K,M] row-major)
__global__ void k_transpose_w_f16(const float* __restrict__ W, _Float16* __restrict__ Wt,
                                  int K, int M)
{
    long long i = blockIdx.x * 256LL + threadIdx.x;
    if (i >= (long long)K * M) return;
    int m = (int)(i / K);
    int k = (int)(i % K);
    Wt[i] = (_Float16)W[(size_t)k * M + m];
}

// act = relu(agg + bias) -> f16 (next layer GEMM input)
__global__ void k_finalize_relu_f16(const float* __restrict__ agg,
                                    const float* __restrict__ bias,
                                    _Float16* __restrict__ act, long long n, int M)
{
    long long i = blockIdx.x * 256LL + threadIdx.x;
    if (i >= n) return;
    float v = agg[i] + bias[i % M];
    v = v > 0.f ? v : 0.f;
    act[i] = (_Float16)v;
}

__global__ void k_add_bias_f32(float* __restrict__ agg, const float* __restrict__ bias,
                               long long n, int M)
{
    long long i = blockIdx.x * 256LL + threadIdx.x;
    if (i < n) agg[i] += bias[i % M];
}

// ---------------------------------------------------------------------------
// Attention logits: al_s[n,h] = sum_c h[n,h,c]*a_src[h,c]  (C == 32 == wave)
// ---------------------------------------------------------------------------
__global__ __launch_bounds__(256)
void k_attn_logits(const float* __restrict__ h, const float* __restrict__ a_src,
                   const float* __restrict__ a_dst, float* __restrict__ al_s,
                   float* __restrict__ al_d, int N, int H)
{
    int wid  = (int)((blockIdx.x * 256LL + threadIdx.x) >> 5);
    int lane = threadIdx.x & 31;
    if (wid >= N * H) return;
    int n  = wid / H;
    int hh = wid % H;
    const int M = H * 32;
    float v = h[(size_t)n * M + hh * 32 + lane];
    float s = v * a_src[hh * 32 + lane];
    float d = v * a_dst[hh * 32 + lane];
    #pragma unroll
    for (int off = 16; off >= 1; off >>= 1) {
        s += __shfl_down(s, off, 32);
        d += __shfl_down(d, off, 32);
    }
    if (lane == 0) { al_s[wid] = s; al_d[wid] = d; }
}

// ---------------------------------------------------------------------------
// Edge kernels. Edges 0..E-1 from edge_index; edges E..E+N-1 are self-loops.
// ---------------------------------------------------------------------------
__device__ __forceinline__ float leaky02(float t) { return t > 0.f ? t : 0.2f * t; }

__device__ __forceinline__ void atomicMaxF32(float* addr, float val)
{
    unsigned int* ua = (unsigned int*)addr;
    unsigned int cur = *ua;
    while (__uint_as_float(cur) < val) {
        unsigned int assumed = cur;
        cur = atomicCAS(ua, assumed, __float_as_uint(val));
        if (cur == assumed) break;
    }
}

__global__ void k_edge_max(const int* __restrict__ src, const int* __restrict__ dst,
                           const float* __restrict__ al_s, const float* __restrict__ al_d,
                           float* __restrict__ emax, long long E, int N, int H)
{
    long long i = blockIdx.x * 256LL + threadIdx.x;
    long long total = (E + N) * H;
    if (i >= total) return;
    long long e = i / H;
    int hh = (int)(i % H);
    int s, d;
    if (e < E) { s = src[e]; d = dst[e]; } else { s = d = (int)(e - E); }
    float t = leaky02(al_s[(size_t)s * H + hh] + al_d[(size_t)d * H + hh]);
    atomicMaxF32(&emax[(size_t)d * H + hh], t);
}

__global__ void k_edge_expsum(const int* __restrict__ src, const int* __restrict__ dst,
                              const float* __restrict__ al_s, const float* __restrict__ al_d,
                              const float* __restrict__ emax, float* __restrict__ denom,
                              long long E, int N, int H)
{
    long long i = blockIdx.x * 256LL + threadIdx.x;
    long long total = (E + N) * H;
    if (i >= total) return;
    long long e = i / H;
    int hh = (int)(i % H);
    int s, d;
    if (e < E) { s = src[e]; d = dst[e]; } else { s = d = (int)(e - E); }
    float t  = leaky02(al_s[(size_t)s * H + hh] + al_d[(size_t)d * H + hh]);
    float ex = __expf(t - emax[(size_t)d * H + hh]);
    atomicAdd(&denom[(size_t)d * H + hh], ex);
}

// One wave per edge; lane = channel. Lanes 0..H-1 compute alpha per head,
// broadcast via shuffle, then 32-wide gather/scatter-add per head.
__global__ __launch_bounds__(256)
void k_edge_scatter(const int* __restrict__ src, const int* __restrict__ dst,
                    const float* __restrict__ al_s, const float* __restrict__ al_d,
                    const float* __restrict__ emax, const float* __restrict__ denom,
                    const float* __restrict__ h, float* __restrict__ out,
                    long long E, int N, int H)
{
    long long wid = (blockIdx.x * 256LL + threadIdx.x) >> 5;
    int lane = threadIdx.x & 31;
    if (wid >= E + N) return;
    int s, d;
    if (wid < E) { s = src[wid]; d = dst[wid]; } else { s = d = (int)(wid - E); }

    float alpha_l = 0.f;
    if (lane < H) {
        float t  = leaky02(al_s[(size_t)s * H + lane] + al_d[(size_t)d * H + lane]);
        float ex = __expf(t - emax[(size_t)d * H + lane]);
        alpha_l  = ex / denom[(size_t)d * H + lane];   // denom>0: self-loops
    }
    const int M = H * 32;
    for (int hh = 0; hh < H; ++hh) {
        float a = __shfl(alpha_l, hh, 32);
        float v = h[(size_t)s * M + hh * 32 + lane] * a;
        atomicAdd(&out[(size_t)d * M + hh * 32 + lane], v);
    }
}

// ---------------------------------------------------------------------------
// Pooling: wave per node (lane = channel, C == 32), then tiny dense layer.
// ---------------------------------------------------------------------------
__global__ __launch_bounds__(256)
void k_pool_accum(const float* __restrict__ h2, const int* __restrict__ batch,
                  float* __restrict__ sums, float* __restrict__ cnts, int N)
{
    long long n = (blockIdx.x * 256LL + threadIdx.x) >> 5;
    int lane = threadIdx.x & 31;
    if (n >= N) return;
    int g = batch[n];
    atomicAdd(&sums[(size_t)g * 32 + lane], h2[(size_t)n * 32 + lane]);
    if (lane == 0) atomicAdd(&cnts[g], 1.0f);
}

__global__ void k_pool_final(const float* __restrict__ sums, const float* __restrict__ cnts,
                             const float* __restrict__ lin_w, const float* __restrict__ lin_b,
                             float* __restrict__ out, int G, int OUT)
{
    int i = (int)(blockIdx.x * 256LL + threadIdx.x);
    if (i >= G * OUT) return;
    int g = i / OUT, o = i % OUT;
    float c = cnts[g];
    c = c > 1.f ? c : 1.f;
    float acc = 0.f;
    #pragma unroll
    for (int k = 0; k < 32; ++k)
        acc += sums[g * 32 + k] * lin_w[k * OUT + o];
    out[i] = acc / c + lin_b[o];
}

// ---------------------------------------------------------------------------
// Host launcher
// ---------------------------------------------------------------------------
static inline unsigned cdivu(long long a, long long b) { return (unsigned)((a + b - 1) / b); }

extern "C" void kernel_launch(void* const* d_in, const int* in_sizes, int n_in,
                              void* d_out, int out_size, void* d_ws, size_t ws_size,
                              hipStream_t stream)
{
    const float* x      = (const float*)d_in[0];
    const int*   eidx   = (const int*)d_in[1];
    const int*   batch  = (const int*)d_in[2];
    const float* Wm[3]   = {(const float*)d_in[3], (const float*)d_in[7],  (const float*)d_in[11]};
    const float* asrc[3] = {(const float*)d_in[4], (const float*)d_in[8],  (const float*)d_in[12]};
    const float* adst[3] = {(const float*)d_in[5], (const float*)d_in[9],  (const float*)d_in[13]};
    const float* bias[3] = {(const float*)d_in[6], (const float*)d_in[10], (const float*)d_in[14]};
    const float* lin_w  = (const float*)d_in[15];
    const float* lin_b  = (const float*)d_in[16];

    const int       N = in_sizes[0] / 128;
    const long long E = in_sizes[1] / 2;
    const int Kd[3] = {128, 256, 256};
    const int Md[3] = {256, 256, 32};
    const int Hd[3] = {8, 8, 1};

    const int* src = eidx;
    const int* dst = eidx + E;

    // workspace carve
    char* p = (char*)d_ws;
    auto carve = [&](size_t bytes) -> char* {
        char* r = p;
        p += (bytes + 255) & ~(size_t)255;
        return r;
    };
    _Float16* act16 = (_Float16*)carve((size_t)N * 256 * 2);
    _Float16* Wt    = (_Float16*)carve((size_t)256 * 256 * 2);
    float* h      = (float*)carve((size_t)N * 256 * 4);
    float* outagg = (float*)carve((size_t)N * 256 * 4);
    float* al_s   = (float*)carve((size_t)N * 8 * 4);
    float* al_d   = (float*)carve((size_t)N * 8 * 4);
    float* emax   = (float*)carve((size_t)N * 8 * 4);
    float* denom  = (float*)carve((size_t)N * 8 * 4);
    float* sums   = (float*)carve((size_t)256 * 32 * 4);
    float* cnts   = (float*)carve((size_t)256 * 4);

    k_f32_to_f16<<<cdivu((long long)N * 128, 256), 256, 0, stream>>>(x, act16, (long long)N * 128);

    for (int L = 0; L < 3; ++L) {
        const int K = Kd[L], M = Md[L], H = Hd[L];
        const long long NM = (long long)N * M;
        const long long NH = (long long)N * H;
        const long long ET = E + N;

        k_transpose_w_f16<<<cdivu((long long)K * M, 256), 256, 0, stream>>>(Wm[L], Wt, K, M);

        dim3 ggrid(cdivu(N, BM), cdivu(M, BN));
        gat_wmma_gemm_f16<<<ggrid, 256, 0, stream>>>(act16, Wt, h, N, K, M);

        k_attn_logits<<<cdivu(NH * 32, 256), 256, 0, stream>>>(h, asrc[L], adst[L],
                                                               al_s, al_d, N, H);

        k_fill_f32<<<cdivu(NH, 256), 256, 0, stream>>>(emax, -3.0e38f, NH);
        k_fill_f32<<<cdivu(NH, 256), 256, 0, stream>>>(denom, 0.0f, NH);
        k_fill_f32<<<cdivu(NM, 256), 256, 0, stream>>>(outagg, 0.0f, NM);

        k_edge_max<<<cdivu(ET * H, 256), 256, 0, stream>>>(src, dst, al_s, al_d, emax, E, N, H);
        k_edge_expsum<<<cdivu(ET * H, 256), 256, 0, stream>>>(src, dst, al_s, al_d, emax,
                                                              denom, E, N, H);
        k_edge_scatter<<<cdivu(ET * 32, 256), 256, 0, stream>>>(src, dst, al_s, al_d, emax,
                                                                denom, h, outagg, E, N, H);

        if (L < 2)
            k_finalize_relu_f16<<<cdivu(NM, 256), 256, 0, stream>>>(outagg, bias[L], act16, NM, M);
        else
            k_add_bias_f32<<<cdivu(NM, 256), 256, 0, stream>>>(outagg, bias[L], NM, M);
    }

    k_fill_f32<<<cdivu(256 * 32, 256), 256, 0, stream>>>(sums, 0.0f, 256 * 32);
    k_fill_f32<<<1, 256, 0, stream>>>(cnts, 0.0f, 256);
    k_pool_accum<<<cdivu((long long)N * 32, 256), 256, 0, stream>>>(outagg, batch, sums, cnts, N);
    k_pool_final<<<cdivu(256 * 64, 256), 256, 0, stream>>>(sums, cnts, lin_w, lin_b,
                                                           (float*)d_out, 256, 64);
}
